// CapsuleNetwork_86766929314213
// MI455X (gfx1250) — matrix operand order repaired and checked by reading, not compiled, not run on hardware
//
#include <hip/hip_runtime.h>

// CapsuleNetwork forward for MI455X (gfx1250, wave32).
// N = 8192 tokens, D = 256 channels, PC=32 caps x PD=8, OC=16 x OD=16.

#define NTOK 8192
#define DCH  256

typedef __bf16 bf16;
typedef __attribute__((ext_vector_type(16))) bf16  v16bf;
typedef __attribute__((ext_vector_type(8)))  float v8f;

__device__ __forceinline__ bf16 f2bf(float f) {
  unsigned u = __builtin_bit_cast(unsigned, f);
  u += 0x7FFFu + ((u >> 16) & 1u);              // round-to-nearest-even
  unsigned short s = (unsigned short)(u >> 16);
  return __builtin_bit_cast(bf16, s);
}

// ---------------------------------------------------------------- prep: fold
// conv1 bias + BN(running stats) into per-channel scale/shift.
__global__ void caps_prep_bn(const float* __restrict__ gamma,
                             const float* __restrict__ beta,
                             const float* __restrict__ mean,
                             const float* __restrict__ var,
                             const float* __restrict__ conv_b,
                             float* __restrict__ sc, float* __restrict__ sh) {
  int c = threadIdx.x;
  float s = gamma[c] * rsqrtf(var[c] + 1e-5f);
  sc[c] = s;
  sh[c] = (conv_b[c] - mean[c]) * s + beta[c];
}

// ------------------------------------------------------- pack x -> bf16 A-frags
// A-frag layout (16-bit A 16x32): lane L (M = L&15, half = L>>4), element e:
//   K = kt*32 + 16*(e>>3) + 8*half + (e&7)
// Fragment storage: ((mt*8 + kt)*32 + lane)*16 + e  (contiguous 32B per lane).
__global__ void caps_pack_x(const float* __restrict__ x, bf16* __restrict__ ax) {
  int tid  = blockIdx.x * 256 + threadIdx.x;   // 512 mt * 8 kt * 32 lanes
  int lane = tid & 31, frag = tid >> 5;
  int mt = frag >> 3, kt = frag & 7;
  int m    = mt * 16 + (lane & 15);
  int half = lane >> 4;
  v16bf v;
#pragma unroll
  for (int e = 0; e < 16; ++e) {
    int k = kt * 32 + 16 * (e >> 3) + 8 * half + (e & 7);
    v[e] = f2bf(x[(size_t)m * DCH + k]);
  }
  *(v16bf*)(ax + (size_t)tid * 16) = v;
}

// ---------------------------------------- pack center-tap weights -> B-frags
// B = W^T so B[k=in_ch][n=out_ch] = w[out*1280 + in*5 + 2].
// B-frag layout (16-bit B 32x16): lane L (N = L&15, half = L>>4), element e:
//   K = kt*32 + 16*(e>>3) + 8*half + (e&7)
__global__ void caps_pack_w(const float* __restrict__ w1,
                            const float* __restrict__ w2,
                            bf16* __restrict__ b1, bf16* __restrict__ b2) {
  int tid  = blockIdx.x * 256 + threadIdx.x;   // 2 * 16 nt * 8 kt * 32 lanes
  int lane = tid & 31, frag = tid >> 5;
  int which = frag >> 7, rem = frag & 127;     // rem = nt*8 + kt
  int nt = rem >> 3, kt = rem & 7;
  const float* w = which ? w2 : w1;
  bf16* dst      = which ? b2 : b1;
  int col  = nt * 16 + (lane & 15);
  int half = lane >> 4;
  v16bf v;
#pragma unroll
  for (int e = 0; e < 16; ++e) {
    int k = kt * 32 + 16 * (e >> 3) + 8 * half + (e & 7);
    v[e] = f2bf(w[(size_t)col * 1280 + (size_t)k * 5 + 2]);
  }
  *(v16bf*)(dst + (size_t)(rem * 32 + lane) * 16) = v;
}

// Stage this block's A fragments (8 kt x 32 lanes x 32B = 8KB, shared by all
// 8 waves / 8 nt tiles) into LDS with gfx1250 async-to-LDS copies.
// IOFFSET applies to both the LDS and global side, so the two 16B halves of a
// lane's 32B fragment slice are the same instruction with offset:16.
__device__ __forceinline__ void stage_a_async(const bf16* __restrict__ ax,
                                              bf16* sA, int mt, int wave,
                                              int lane) {
  unsigned lds = (unsigned)(size_t)&sA[(wave * 32 + lane) * 16];
  const bf16* g = ax + (size_t)((mt * 8 + wave) * 32 + lane) * 16;
  asm volatile("global_load_async_to_lds_b128 %0, %1, off"
               :: "v"(lds), "v"(g) : "memory");
  asm volatile("global_load_async_to_lds_b128 %0, %1, off offset:16"
               :: "v"(lds), "v"(g) : "memory");
  asm volatile("s_wait_asynccnt 0x0" ::: "memory");
  __syncthreads();
}

// --------------------------------------------- GEMM1: h = relu(bn(x@W1^T+b1))
// Block = 8 waves sharing one mt (A staged once in LDS), wave = one 16x16
// output tile, K = 256 in 8 WMMA steps. Epilogue writes h directly in bf16
// A-fragment layout for GEMM2.
__global__ __launch_bounds__(256) void caps_gemm1(
    const bf16* __restrict__ ax, const bf16* __restrict__ b1,
    const float* __restrict__ sc, const float* __restrict__ sh,
    bf16* __restrict__ ha) {
  __shared__ bf16 sA[8 * 32 * 16];             // 8 KB
  int wave = threadIdx.x >> 5, lane = threadIdx.x & 31;
  int mt = blockIdx.x >> 1;
  int nt = (blockIdx.x & 1) * 8 + wave;
  stage_a_async(ax, sA, mt, wave, lane);

  v8f acc = {0.f, 0.f, 0.f, 0.f, 0.f, 0.f, 0.f, 0.f};
#pragma unroll
  for (int kt = 0; kt < 8; ++kt) {
    v16bf a = *(const v16bf*)(sA + (size_t)(kt * 32 + lane) * 16);
    v16bf b = *(const v16bf*)(b1 + (size_t)((nt * 8 + kt) * 32 + lane) * 16);
    acc = __builtin_amdgcn_wmma_f32_16x16x32_bf16(false, a, false, b,
                                                  (short)0, acc, false, false);
  }
  // C/D layout: lane L, row r -> M = r + 8*(L>>4), N = L&15
  int half = lane >> 4;
  int col  = nt * 16 + (lane & 15);
  float s = sc[col], t = sh[col];
  int kt2 = col >> 5, kl = col & 31;
  int e2  = 8 * (kl >> 4) + (kl & 7);
  int l2h = ((kl >> 3) & 1) << 4;
#pragma unroll
  for (int r = 0; r < 8; ++r) {
    int m   = mt * 16 + r + 8 * half;
    float h = fmaxf(acc[r] * s + t, 0.0f);
    ha[(size_t)((mt * 8 + kt2) * 32 + (l2h | (m & 15))) * 16 + e2] = f2bf(h);
  }
}

// -------------------------------------------------- GEMM2: p = h@W2^T + b2
__global__ __launch_bounds__(256) void caps_gemm2(
    const bf16* __restrict__ ha, const bf16* __restrict__ b2,
    const float* __restrict__ pcb, float* __restrict__ p) {
  __shared__ bf16 sA[8 * 32 * 16];
  int wave = threadIdx.x >> 5, lane = threadIdx.x & 31;
  int mt = blockIdx.x >> 1;
  int nt = (blockIdx.x & 1) * 8 + wave;
  stage_a_async(ha, sA, mt, wave, lane);

  v8f acc = {0.f, 0.f, 0.f, 0.f, 0.f, 0.f, 0.f, 0.f};
#pragma unroll
  for (int kt = 0; kt < 8; ++kt) {
    v16bf a = *(const v16bf*)(sA + (size_t)(kt * 32 + lane) * 16);
    v16bf b = *(const v16bf*)(b2 + (size_t)((nt * 8 + kt) * 32 + lane) * 16);
    acc = __builtin_amdgcn_wmma_f32_16x16x32_bf16(false, a, false, b,
                                                  (short)0, acc, false, false);
  }
  int half = lane >> 4;
  int col  = nt * 16 + (lane & 15);
  float bb = pcb[col];
#pragma unroll
  for (int r = 0; r < 8; ++r) {
    int m = mt * 16 + r + 8 * half;
    p[(size_t)m * 256 + col] = acc[r] + bb;
  }
}

// ------------------------------------------------ squash + dynamic routing
// Wave = one token, lane = in-capsule i (PD=8 floats lane-local -> squash is
// lane-local). u_hat[j,i,o] (32KB fp32/token) lives in LDS with stride-33
// padding; 8 tokens/block -> 288KB of the 320KB WGP LDS. Keeping u_hat in LDS
// avoids a 256MB HBM round-trip for the routing iterations.
#define U_STR 33
#define U_SZ  (256 * U_STR)          // floats per token

__global__ __launch_bounds__(256) void caps_routing(
    const float* __restrict__ p, const float* __restrict__ W,
    float* __restrict__ out) {
  extern __shared__ float smem[];
  int wave = threadIdx.x >> 5, lane = threadIdx.x & 31;
  int n = blockIdx.x * 8 + wave;
  float* uw = smem + wave * U_SZ;                       // u_hat[j*16+o][i]
  float* cw = smem + 8 * U_SZ + wave * 512;             // c[j][i]
  float* vw = smem + 8 * U_SZ + 8 * 512 + wave * 256;   // v[j][o]

  // load p[n, lane*8 .. +7] and squash (capsule = lane)
  const float4* pr = (const float4*)(p + (size_t)n * 256 + lane * 8);
  float4 pa = pr[0], pb = pr[1];
  float sq = pa.x * pa.x + pa.y * pa.y + pa.z * pa.z + pa.w * pa.w +
             pb.x * pb.x + pb.y * pb.y + pb.z * pb.z + pb.w * pb.w;
  float scl = (sq / (1.0f + sq)) * rsqrtf(sq + 1e-8f);
  pa.x *= scl; pa.y *= scl; pa.z *= scl; pa.w *= scl;
  pb.x *= scl; pb.y *= scl; pb.z *= scl; pb.w *= scl;

  // u_hat[j, lane, o] = W[j,lane,o,:] . p[lane,:]
  for (int j = 0; j < 16; ++j) {
    if (j < 15)
      __builtin_prefetch(W + (size_t)((j + 1) * 32 + lane) * 128, 0, 0);
    const float4* w4 = (const float4*)(W + (size_t)(j * 32 + lane) * 128);
#pragma unroll
    for (int o = 0; o < 16; ++o) {
      float4 wa = w4[2 * o], wb = w4[2 * o + 1];
      float a = wa.x * pa.x + wa.y * pa.y + wa.z * pa.z + wa.w * pa.w +
                wb.x * pb.x + wb.y * pb.y + wb.z * pb.z + wb.w * pb.w;
      uw[(j * 16 + o) * U_STR + lane] = a;
    }
  }

  float blog[16];
#pragma unroll
  for (int j = 0; j < 16; ++j) blog[j] = 0.0f;
  int jj = lane >> 1, ob = (lane & 1) * 8;   // sv ownership: lane -> (j, o-half)
  __syncthreads();

  for (int t = 0; t < 3; ++t) {
    // softmax over j (lane-local), publish c[j][i]
    float mx = blog[0];
#pragma unroll
    for (int j = 1; j < 16; ++j) mx = fmaxf(mx, blog[j]);
    float ce[16], esum = 0.0f;
#pragma unroll
    for (int j = 0; j < 16; ++j) { ce[j] = __expf(blog[j] - mx); esum += ce[j]; }
    float inv = 1.0f / esum;
#pragma unroll
    for (int j = 0; j < 16; ++j) cw[j * 32 + lane] = ce[j] * inv;
    __syncthreads();

    // sv[jj, ob..ob+7] = sum_i c[jj,i] * u_hat[jj,i,o]
    float sv[8];
#pragma unroll
    for (int o = 0; o < 8; ++o) sv[o] = 0.0f;
    for (int i = 0; i < 32; ++i) {
      float c = cw[jj * 32 + i];
#pragma unroll
      for (int o = 0; o < 8; ++o)
        sv[o] += c * uw[(jj * 16 + ob + o) * U_STR + i];
    }
    // squash over full 16-o row: combine the two half-rows with partner lane
    float s2 = 0.0f;
#pragma unroll
    for (int o = 0; o < 8; ++o) s2 += sv[o] * sv[o];
    s2 += __shfl_xor(s2, 1, 32);
    float vscl = (s2 / (1.0f + s2)) * rsqrtf(s2 + 1e-8f);

    if (t < 2) {
      #pragma unroll
      for (int o = 0; o < 8; ++o) vw[jj * 16 + ob + o] = sv[o] * vscl;
      __syncthreads();
      // blog[j,lane] += sum_o u_hat[j,lane,o] * v[j,o]
#pragma unroll
      for (int j = 0; j < 16; ++j) {
        float acc = 0.0f;
#pragma unroll
        for (int o = 0; o < 16; ++o)
          acc += uw[(j * 16 + o) * U_STR + lane] * vw[j * 16 + o];
        blog[j] += acc;
      }
      __syncthreads();
    } else {
      // out[n, o*16 + j] = v[n, j, o]   (swapaxes(1,2) + flatten)
#pragma unroll
      for (int o = 0; o < 8; ++o)
        out[(size_t)n * 256 + (ob + o) * 16 + jj] = sv[o] * vscl;
    }
  }
}

// ------------------------------------------------------------------ launcher
extern "C" void kernel_launch(void* const* d_in, const int* in_sizes, int n_in,
                              void* d_out, int out_size, void* d_ws, size_t ws_size,
                              hipStream_t stream) {
  (void)in_sizes; (void)n_in; (void)out_size; (void)ws_size;
  const float* x       = (const float*)d_in[0];
  const float* conv1_w = (const float*)d_in[1];
  const float* conv1_b = (const float*)d_in[2];
  const float* gamma   = (const float*)d_in[3];
  const float* beta    = (const float*)d_in[4];
  const float* mean    = (const float*)d_in[5];
  const float* var     = (const float*)d_in[6];
  const float* pc_w    = (const float*)d_in[7];
  const float* pc_b    = (const float*)d_in[8];
  const float* W_route = (const float*)d_in[9];
  float* out = (float*)d_out;
  char*  ws  = (char*)d_ws;

  // workspace layout (bytes)
  bf16*  ax = (bf16*)(ws + 0);                    // 512*8*512 bf16 = 4 MiB
  bf16*  b1 = (bf16*)(ws + 4194304);              // 128 KiB
  bf16*  b2 = (bf16*)(ws + 4325376);              // 128 KiB
  bf16*  ha = (bf16*)(ws + 4456448);              // 4 MiB
  float* pp = (float*)(ws + 8650752);             // 8192*256 f32 = 8 MiB
  float* sc = (float*)(ws + 17039360);            // 1 KiB
  float* sh = (float*)(ws + 17040384);            // 1 KiB

  caps_prep_bn<<<1, 256, 0, stream>>>(gamma, beta, mean, var, conv1_b, sc, sh);
  caps_pack_x<<<512, 256, 0, stream>>>(x, ax);
  caps_pack_w<<<32, 256, 0, stream>>>(conv1_w, pc_w, b1, b2);
  caps_gemm1<<<1024, 256, 0, stream>>>(ax, b1, sc, sh, ha);
  caps_gemm2<<<1024, 256, 0, stream>>>(ha, b2, pc_b, pp);
  caps_routing<<<1024, 256, (8 * U_SZ + 8 * 512 + 8 * 256) * sizeof(float),
                 stream>>>(pp, W_route, out);
}
